// SlidingWindowPattern_55490977465134
// MI455X (gfx1250) — matrix operand order repaired and checked by reading, not compiled, not run on hardware
//
#include <hip/hip_runtime.h>

// ---- problem constants (match reference) ----
#define BB   4
#define SS   4096
#define DD   1024
#define EE   64          // IDX (q/k head dim)
#define WW   128         // window
#define BS   (BB * SS)   // 16384 total rows
#define WROW 40          // padded LDS row stride (halves) for bank-conflict-free b128

typedef _Float16 v16h __attribute__((ext_vector_type(16)));
typedef _Float16 h8   __attribute__((ext_vector_type(8)));
typedef _Float16 h4   __attribute__((ext_vector_type(4)));
typedef float    v8f  __attribute__((ext_vector_type(8)));
typedef float    f4   __attribute__((ext_vector_type(4)));

__device__ __forceinline__ v8f wmma_f16(v16h a, v16h b, v8f c) {
  // D = A(16x32 f16) * B(32x16 f16) + C(16x16 f32)
  return __builtin_amdgcn_wmma_f32_16x16x32_f16(
      /*neg_a=*/false, a, /*neg_b=*/false, b,
      /*c_mod=*/(short)0, c, /*reuse_a=*/false, /*reuse_b=*/false);
}

// ---------------------------------------------------------------------------
// Kernel 0: one-shot f32 -> f16 conversion of wq/wk (64x1024 each).
// ---------------------------------------------------------------------------
__global__ void __launch_bounds__(256)
swp_cvt_weights(const float* __restrict__ wq, const float* __restrict__ wk,
                _Float16* __restrict__ wqh, _Float16* __restrict__ wkh) {
  const int i = blockIdx.x * blockDim.x + threadIdx.x;   // [0, EE*DD/4)
  if (i >= EE * DD / 4) return;
  f4 q = ((const f4*)wq)[i];
  f4 k = ((const f4*)wk)[i];
  h4 qo, ko;
#pragma unroll
  for (int t = 0; t < 4; ++t) { qo[t] = (_Float16)q[t]; ko[t] = (_Float16)k[t]; }
  ((h4*)wqh)[i] = qo;
  ((h4*)wkh)[i] = ko;
}

// ---------------------------------------------------------------------------
// Kernel 1: stream -inf into the whole mask (256 MB) with non-temporal b128.
// ---------------------------------------------------------------------------
__global__ void __launch_bounds__(256)
swp_fill_neginf(f4* __restrict__ out, int n4) {
  const float ninf = -__builtin_inff();
  f4 v; v[0] = ninf; v[1] = ninf; v[2] = ninf; v[3] = ninf;
  size_t i = (size_t)blockIdx.x * blockDim.x + threadIdx.x;
  const size_t stride = (size_t)gridDim.x * blockDim.x;
  for (; i < (size_t)n4; i += stride)
    __builtin_nontemporal_store(v, out + i);
}

// ---------------------------------------------------------------------------
// Kernel 2: Q/K projections via WMMA f16.
// 256 threads = 8 waves; each wave owns one 16-row tile of x.
// The 8 KB weight k-slice (64e x 32d x f16 x {wq,wk}) is cooperatively staged
// into LDS, double-buffered (stage kb+1 while computing kb).  B tiles are then
// short-latency ds_load_b128 pairs instead of L2 round-trips, and global
// weight traffic drops 8x.  LDS rows padded to 80 B for conflict-free reads.
// ---------------------------------------------------------------------------
__global__ void __launch_bounds__(256)
swp_proj_qk(const float* __restrict__ x, const _Float16* __restrict__ wqh,
            const _Float16* __restrict__ wkh, _Float16* __restrict__ qh,
            _Float16* __restrict__ kh) {
  __shared__ _Float16 sw[2][2][EE * WROW];     // [buf][mat][64 x 40 halves] = 20 KB
  const int tid  = threadIdx.x;
  const int wave = tid >> 5;
  const int lane = tid & 31;
  const int hi = lane >> 4, lp = lane & 15;
  const int grow = blockIdx.x * 128 + wave * 16;   // this wave's 16-row tile

  // staging role: thread -> (row e, 8-half chunk) of the k-slice
  const int se = tid >> 2, sch = (tid & 3) * 8;

  v8f accq[4], acck[4];
#pragma unroll
  for (int et = 0; et < 4; ++et)
#pragma unroll
    for (int r = 0; r < 8; ++r) { accq[et][r] = 0.f; acck[et][r] = 0.f; }

  const float* xrow = x + (size_t)(grow + lp) * DD;

  // stage k-slice 0 into buffer 0
  {
    h8 vq = *(const h8*)(wqh + (size_t)se * DD + sch);
    h8 vk = *(const h8*)(wkh + (size_t)se * DD + sch);
    *(h8*)(&sw[0][0][se * WROW + sch]) = vq;
    *(h8*)(&sw[0][1][se * WROW + sch]) = vk;
  }
  __syncthreads();

  for (int kb = 0; kb < 32; ++kb) {
    const int cur = kb & 1, nxt = cur ^ 1;
    const int dbase = kb * 32;
    // ---- stage next k-slice into the other buffer (global -> LDS) ----
    if (kb + 1 < 32) {
      h8 vq = *(const h8*)(wqh + (size_t)se * DD + (dbase + 32) + sch);
      h8 vk = *(const h8*)(wkh + (size_t)se * DD + (dbase + 32) + sch);
      *(h8*)(&sw[nxt][0][se * WROW + sch]) = vq;
      *(h8*)(&sw[nxt][1][se * WROW + sch]) = vk;
    }
    // ---- A tile: 16x32 f32 from x, cvt to f16 in regs (x is read once) ----
    f4 a0 = *(const f4*)(xrow + dbase + hi * 8);
    f4 a1 = *(const f4*)(xrow + dbase + hi * 8 + 4);
    f4 a2 = *(const f4*)(xrow + dbase + 16 + hi * 8);
    f4 a3 = *(const f4*)(xrow + dbase + 16 + hi * 8 + 4);
    v16h a;   // halves 0..7 -> K = dbase + hi*8 + t ; halves 8..15 -> +16
#pragma unroll
    for (int t = 0; t < 4; ++t) {
      a[t]      = (_Float16)a0[t];
      a[t + 4]  = (_Float16)a1[t];
      a[t + 8]  = (_Float16)a2[t];
      a[t + 12] = (_Float16)a3[t];
    }
    // ---- B tiles from LDS (halves 0..7 -> K = dbase + hi*8 + h ; halves 8..15 -> +16, col = lp) ----
#pragma unroll
    for (int et = 0; et < 4; ++et) {
      const int e = et * 16 + lp;
      v16h bq, bk;
      {
        h8 q0 = *(const h8*)(&sw[cur][0][e * WROW + hi * 8]);
        h8 q1 = *(const h8*)(&sw[cur][0][e * WROW + 16 + hi * 8]);
        h8 k0 = *(const h8*)(&sw[cur][1][e * WROW + hi * 8]);
        h8 k1 = *(const h8*)(&sw[cur][1][e * WROW + 16 + hi * 8]);
#pragma unroll
        for (int t = 0; t < 8; ++t) { bq[t] = q0[t]; bq[t + 8] = q1[t]; bk[t] = k0[t]; bk[t + 8] = k1[t]; }
      }
      accq[et] = wmma_f16(a, bq, accq[et]);
      acck[et] = wmma_f16(a, bk, acck[et]);
    }
    __syncthreads();   // compute(cur) done by all waves; stage(nxt) complete
  }

  // D layout: vgpr r -> row m = r + hi*8, col = lp
#pragma unroll
  for (int et = 0; et < 4; ++et)
#pragma unroll
    for (int r = 0; r < 8; ++r) {
      const int m = r + hi * 8;
      qh[(size_t)(grow + m) * EE + et * 16 + lp] = (_Float16)accq[et][r];
      kh[(size_t)(grow + m) * EE + et * 16 + lp] = (_Float16)acck[et][r];
    }
}

// ---------------------------------------------------------------------------
// Kernel 3: windowed scores (WMMA, wave 0) + stable top-k rank + scatter 0s.
// Block = 128 threads (4 waves) per 16-query tile.  LDS holds the relu'd
// 16 x 144 key-union panel (keys qbase-128 .. qbase+15).
// ---------------------------------------------------------------------------
__global__ void __launch_bounds__(128)
swp_score_select(const _Float16* __restrict__ qh, const _Float16* __restrict__ kh,
                 float* __restrict__ out) {
  __shared__ float sc[16 * 144];
  const int tid = threadIdx.x;
  const int grow = blockIdx.x * 16;            // [0, BS), tiles never cross batch
  const int b = grow / SS;
  const int qbase = grow - b * SS;
  const float ninf = -__builtin_inff();

  if (tid < 32) {                              // wave 0: full EXEC for WMMA
    const int lane = tid, hi = lane >> 4, lp = lane & 15;
    // A: 16 queries x 64 dims (two 16x32 k-steps)
    v16h aq[2];
#pragma unroll
    for (int ks = 0; ks < 2; ++ks) {
      const _Float16* qr = qh + (size_t)(grow + lp) * EE + ks * 32;
      h8 lo = *(const h8*)(qr + hi * 8);
      h8 hh = *(const h8*)(qr + 16 + hi * 8);
#pragma unroll
      for (int t = 0; t < 8; ++t) { aq[ks][t] = lo[t]; aq[ks][t + 8] = hh[t]; }
    }
    for (int kt = 0; kt < 9; ++kt) {           // 9 key tiles = 144 keys
      const int key = qbase - 128 + kt * 16 + lp;
      const _Float16* kr = kh + (size_t)(b * SS + (key < 0 ? 0 : key)) * EE;
      v16h bk0, bk1;
      {
        h8 p0 = *(const h8*)(kr + hi * 8);
        h8 p1 = *(const h8*)(kr + 16 + hi * 8);
        h8 p2 = *(const h8*)(kr + 32 + hi * 8);
        h8 p3 = *(const h8*)(kr + 48 + hi * 8);
#pragma unroll
        for (int t = 0; t < 8; ++t) { bk0[t] = p0[t]; bk0[t + 8] = p1[t]; bk1[t] = p2[t]; bk1[t + 8] = p3[t]; }
      }
      v8f acc;
#pragma unroll
      for (int r = 0; r < 8; ++r) acc[r] = 0.f;
      acc = wmma_f16(aq[0], bk0, acc);
      acc = wmma_f16(aq[1], bk1, acc);
#pragma unroll
      for (int r = 0; r < 8; ++r) {
        const float v = acc[r];
        sc[(r + hi * 8) * 144 + kt * 16 + lp] = v > 0.f ? v : 0.f;  // relu
      }
    }
  }
  __syncthreads();

  // Stable-descending rank selection: rank(j) = #{h: v_h > v_j} +
  // #{h < j: v_h == v_j}; select if rank < ksel (matches stable argsort).
  const int wave = tid >> 5, lane = tid & 31;
  for (int i = wave; i < 16; i += 4) {
    const int qpos = qbase + i;
    const int winlen = (qpos + 1 < WW) ? qpos + 1 : WW;
    int ksel = winlen >> 1; if (ksel < 1) ksel = 1;
    const float* row = &sc[i * 144 + i + 1];   // window offset j -> key qpos-127+j
    const int inval = 127 - qpos;              // j < inval => key < 0 => -inf

    float v[4]; int cnt[4];
#pragma unroll
    for (int c = 0; c < 4; ++c) {
      const int j = lane + 32 * c;
      v[c] = (j < inval) ? ninf : row[j];
      cnt[c] = 0;
    }
    for (int h = 0; h < WW; ++h) {
      const float vh = (h < inval) ? ninf : row[h];
#pragma unroll
      for (int c = 0; c < 4; ++c) {
        const int j = lane + 32 * c;
        cnt[c] += ((vh > v[c]) || (vh == v[c] && h < j)) ? 1 : 0;
      }
    }
#pragma unroll
    for (int c = 0; c < 4; ++c) {
      const int j = lane + 32 * c;
      if (cnt[c] < ksel && j >= inval) {
        const int key = qpos - 127 + j;
        out[(size_t)(b * SS + qpos) * SS + key] = 0.0f;
      }
    }
  }
}

// ---------------------------------------------------------------------------
extern "C" void kernel_launch(void* const* d_in, const int* in_sizes, int n_in,
                              void* d_out, int out_size, void* d_ws, size_t ws_size,
                              hipStream_t stream) {
  const float* x  = (const float*)d_in[0];
  const float* wq = (const float*)d_in[1];
  const float* wk = (const float*)d_in[2];
  float* out = (float*)d_out;

  _Float16* qh  = (_Float16*)d_ws;                   // [BS, 64] f16 (2 MB)
  _Float16* kh  = qh + (size_t)BS * EE;              // [BS, 64] f16 (2 MB)
  _Float16* wqh = kh + (size_t)BS * EE;              // [64, 1024] f16 (128 KB)
  _Float16* wkh = wqh + (size_t)EE * DD;             // [64, 1024] f16 (128 KB)

  // 0) convert weights to f16 once (tiny)
  swp_cvt_weights<<<(EE * DD / 4 + 255) / 256, 256, 0, stream>>>(wq, wk, wqh, wkh);
  // 1) stream -inf mask fill (bandwidth-bound: 256 MB)
  swp_fill_neginf<<<2048, 256, 0, stream>>>((f4*)out, out_size / 4);
  // 2) Q/K projections via WMMA f16 (8 waves/block, LDS-staged weights)
  swp_proj_qk<<<BS / 128, 256, 0, stream>>>(x, wqh, wkh, qh, kh);
  // 3) windowed scores + stable top-k + scatter zeros
  swp_score_select<<<BS / 16, 128, 0, stream>>>(qh, kh, out);
}
